// GCN_surface_4398046511588
// MI455X (gfx1250) — matrix operand order repaired and checked by reading, not compiled, not run on hardware
//
#include <hip/hip_runtime.h>
#include <hip/hip_bf16.h>
#include <cmath>

// ---------------- problem constants (from reference) ----------------
#define NN 50000
#define EE 1600000
#define FF 128
#define DD 256
#define GG 64
#define KK 10

typedef __attribute__((ext_vector_type(16))) _Float16 v16h;
typedef __attribute__((ext_vector_type(8)))  _Float16 v8h;
typedef __attribute__((ext_vector_type(4)))  _Float16 v4h;
typedef __attribute__((ext_vector_type(8)))  float    v8f;

// ---------------- degree / dis ----------------
__global__ void deg_init_kernel(float* dis, int n) {
    int i = blockIdx.x * blockDim.x + threadIdx.x;
    if (i < n) dis[i] = 1.0f;  // self-loop
}

__global__ void deg_accum_kernel(float* dis, const long long* __restrict__ ei, int e) {
    int i = blockIdx.x * blockDim.x + threadIdx.x;
    if (i < e) {
        long long d = ei[(long long)e + i];   // row 1 = dst
        atomicAdd(&dis[d], 1.0f);
    }
}

__global__ void deg_finalize_kernel(float* dis, int n) {
    int i = blockIdx.x * blockDim.x + threadIdx.x;
    if (i < n) dis[i] = rsqrtf(dis[i]);
}

// ---------------- f32 -> f16 convert (vectorized), for layer-0 input x ----------------
__global__ void cvt16_kernel(const float* __restrict__ src, _Float16* __restrict__ dst,
                             int n4) {   // n4 = count/4
    int i = blockIdx.x * blockDim.x + threadIdx.x;
    if (i >= n4) return;
    float4 v = ((const float4*)src)[i];
    v4h h; h[0] = (_Float16)v.x; h[1] = (_Float16)v.y;
    h[2] = (_Float16)v.z; h[3] = (_Float16)v.w;
    ((v4h*)dst)[i] = h;
}

// ---------------- weight pre-swizzle into per-lane B fragments (k = 16*half + h) ----------------
// Bs[((kt*16 + ctile)*32 + lane)*16 + h] = (f16) W[(kt*32 + (lane&16) + h) * 256
//                                                  + ctile*16 + (lane&15)]
// so a B fragment (32x16 f16) is 16 contiguous halves (32B) per lane.
__global__ void wswz_kernel(const float* __restrict__ W, _Float16* __restrict__ Bs, int K) {
    int i = blockIdx.x * blockDim.x + threadIdx.x;
    if (i >= K * DD) return;
    int h    = i & 15;
    int lane = (i >> 4) & 31;
    int c    = (i >> 9) & 15;         // DD/16 = 16 column tiles
    int kt   = i >> 13;               // 16*32*16 = 8192 halves per K-tile of 32
    int k    = kt * 32 + (lane & 16) + h;
    int n    = c * 16 + (lane & 15);
    Bs[i] = (_Float16)W[(long long)k * DD + n];
}

// ---------------- WMMA GEMM: hlin[M,256] = A16[M,K] @ W[K,256] ----------------
// block tile: 128 (M) x 128 (N), 256 threads = 8 wave32
// wave w computes rows [w*16, w*16+16) x all 128 cols -> 8 accumulators
// A (f16) staged in LDS; B fragments pre-swizzled in global (L2 resident) and
// software-pipelined: B for step kt+1 is loaded while the WMMA chain for kt runs.
#define TM 128
#define TN 128
#define TK 32
#define LPAD 8   // pad in halves (16B): keeps vector alignment, spreads banks

template<int K>
__global__ __launch_bounds__(256) void gemm_wmma_kernel(
    const _Float16* __restrict__ A16, const _Float16* __restrict__ Bs,
    float* __restrict__ hlin, int M)
{
    __shared__ _Float16 Ah[TM][TK + LPAD];

    const int tid  = threadIdx.x;
    const int lane = tid & 31;
    const int wave = tid >> 5;
    const int half = (lane >> 4) & 1;       // 0: lanes 0-15, 1: lanes 16-31
    const int rowBase = blockIdx.x * TM;
    const int colBase = blockIdx.y * TN;
    const int ct0 = colBase >> 4;           // first global column tile

    // per-lane B base: fragment (kt,c) lives at 32B granularity
    const _Float16* bbase = Bs + ((((long long)ct0) * 32 + lane) << 4);
    // fragment (kt,c) offset in halves: ((kt*16 + c)*32)*16
    auto loadB = [&](int kt, int c) -> v16h {
        return *(const v16h*)(bbase + ((long long)(kt * 16 + c) << 9));
    };

    v8f acc[8];
#pragma unroll
    for (int c = 0; c < 8; ++c)
#pragma unroll
        for (int r = 0; r < 8; ++r) acc[c][r] = 0.0f;

    constexpr int NSTEP = K / TK;

    // prologue: preload B fragments for kt = 0
    v16h bfrag[8];
#pragma unroll
    for (int c = 0; c < 8; ++c) bfrag[c] = loadB(0, c);

#pragma unroll
    for (int kt = 0; kt < NSTEP; ++kt) {
        const int k0 = kt * TK;
        // stage A tile: 128 rows x 32 halves = 8KB; 512 x 16B chunks, 2 per thread.
        // Row clamp instead of zero-fill: clamped rows only affect C rows >= M,
        // which are never stored -> no EXEC-masked guard in the hot loop.
#pragma unroll
        for (int it = 0; it < 2; ++it) {
            int i  = it * 256 + tid;        // 16B-chunk index
            int r  = i >> 2;                // 4 chunks per row
            int c8 = i & 3;
            int gr = rowBase + r;
            gr = (gr < M) ? gr : (M - 1);
            v8h v = *(const v8h*)(A16 + (long long)gr * K + k0 + c8 * 8);
            *(v8h*)&Ah[r][c8 * 8] = v;
        }
        __syncthreads();

        // A fragment (16x32 f16): lanes 0-15 K{0..15}; lanes 16-31 K{16..31}
        v16h afrag;
        {
            int m = wave * 16 + (lane & 15);
            const v8h* arow = (const v8h*)Ah[m];
            v8h lo = arow[2 * half];
            v8h hi = arow[2 * half + 1];
            afrag = __builtin_shufflevector(lo, hi,
                0,1,2,3,4,5,6,7,8,9,10,11,12,13,14,15);
        }

        // issue next step's B loads; they complete during this WMMA chain + barrier
        v16h bnext[8];
        if (kt + 1 < NSTEP) {
#pragma unroll
            for (int c = 0; c < 8; ++c) bnext[c] = loadB(kt + 1, c);
        }

#pragma unroll
        for (int c = 0; c < 8; ++c)
            acc[c] = __builtin_amdgcn_wmma_f32_16x16x32_f16(
                false, afrag, false, bfrag[c], (short)0, acc[c], false, false);

        __syncthreads();

        if (kt + 1 < NSTEP) {
#pragma unroll
            for (int c = 0; c < 8; ++c) bfrag[c] = bnext[c];   // SSA rename (unrolled)
        }
    }

    // store: VGPR r -> row (lane<16 ? r : r+8), col = lane%16
    // single 64-bit base, all stores at compile-time immediate offsets
    const int mbase = rowBase + wave * 16 + half * 8;
    float* base = hlin + (long long)mbase * DD + colBase + (lane & 15);
    if (rowBase + TM <= M) {                // uniform fast path (no per-lane guard)
#pragma unroll
        for (int c = 0; c < 8; ++c)
#pragma unroll
            for (int r = 0; r < 8; ++r)
                base[r * DD + c * 16] = acc[c][r];
    } else {
#pragma unroll
        for (int c = 0; c < 8; ++c)
#pragma unroll
            for (int r = 0; r < 8; ++r)
                if (mbase + r < M) base[r * DD + c * 16] = acc[c][r];
    }
}

// ---------------- agg init: agg = hlin * dis[row]^2 (self loop) ----------------
__global__ void agg_init_kernel(const float* __restrict__ hlin, float* __restrict__ agg,
                                const float* __restrict__ dis, int m) {
    int i = blockIdx.x * blockDim.x + threadIdx.x;
    if (i < m * DD) {
        int row = i >> 8;  // DD == 256
        float ds = dis[row];
        agg[i] = hlin[i] * ds * ds;
    }
}

// ---------------- edge scatter: one wave32 per edge, 8 channels/lane ----------------
__global__ __launch_bounds__(256) void scatter_kernel(
    const float* __restrict__ hlin, float* __restrict__ agg,
    const float* __restrict__ dis, const long long* __restrict__ ei, int e)
{
    int lane = threadIdx.x & 31;
    int gwave = blockIdx.x * 8 + (threadIdx.x >> 5);
    if (gwave >= e) return;
    long long s = ei[gwave];                 // row 0 = src
    long long d = ei[(long long)e + gwave];  // row 1 = dst
    float norm = dis[s] * dis[d];
    const float* hs = hlin + s * DD;
    float* ad = agg + d * DD;
#pragma unroll
    for (int i = 0; i < 8; ++i) {
        int c = i * 32 + lane;
        atomicAdd(&ad[c], hs[c] * norm);
    }
}

// ---------------- bias + tanh, in place; also emit f16 copy for next GEMM ----------------
__global__ void bias_tanh_kernel(float* __restrict__ agg, _Float16* __restrict__ h16,
                                 const float* __restrict__ b, int m) {
    int i = blockIdx.x * blockDim.x + threadIdx.x;
    if (i < m * DD) {
        float v = tanhf(agg[i] + b[i & (DD - 1)]);
        agg[i] = v;
        h16[i] = (_Float16)v;
    }
}

// ---------------- pooling ----------------
__device__ inline void atomicMaxF(float* addr, float val) {
    if (val >= 0.0f) atomicMax((int*)addr, __float_as_int(val));
    else             atomicMin((unsigned int*)addr, __float_as_uint(val));
}

__global__ void pool_init_kernel(float* gmax, float* gsum, float* cnt) {
    int i = blockIdx.x * blockDim.x + threadIdx.x;
    if (i < GG * DD) gmax[i] = -INFINITY;
    else if (i < 2 * GG * DD) gsum[i - GG * DD] = 0.0f;
    else if (i < 2 * GG * DD + GG) cnt[i - 2 * GG * DD] = 0.0f;
}

__global__ void pool_kernel(const float* __restrict__ h, const long long* __restrict__ batch,
                            float* gmax, float* gsum, float* cnt) {
    int n = blockIdx.x;         // one block (256 threads) per node
    int d = threadIdx.x;
    long long b = batch[n];
    float v = h[(long long)n * DD + d];
    atomicAdd(&gsum[b * DD + d], v);
    atomicMaxF(&gmax[b * DD + d], v);
    if (d == 0) atomicAdd(&cnt[b], 1.0f);
}

__global__ void build_g_kernel(const float* __restrict__ gmax, const float* __restrict__ gsum,
                               const float* __restrict__ cnt, float* __restrict__ g) {
    int i = blockIdx.x * blockDim.x + threadIdx.x;
    if (i >= GG * 2 * DD) return;
    int gi = i / (2 * DD), j = i % (2 * DD);
    float v;
    if (j < DD) v = gmax[gi * DD + j];
    else        v = gsum[gi * DD + (j - DD)] / fmaxf(cnt[gi], 1.0f);
    g[i] = v;
}

// ---------------- small FC layers (G=64 rows, negligible cost) ----------------
__global__ void fc_kernel(const float* __restrict__ X, const float* __restrict__ Wm,
                          const float* __restrict__ bias, float* __restrict__ Y,
                          int M, int K, int Nout, int relu) {
    int i = blockIdx.x * blockDim.x + threadIdx.x;
    if (i >= M * Nout) return;
    int m = i / Nout, n = i % Nout;
    float acc = bias[n];
    for (int k = 0; k < K; ++k) acc += X[m * K + k] * Wm[k * Nout + n];
    Y[i] = relu ? fmaxf(acc, 0.0f) : acc;
}

// ---------------- host orchestration ----------------
extern "C" void kernel_launch(void* const* d_in, const int* in_sizes, int n_in,
                              void* d_out, int out_size, void* d_ws, size_t ws_size,
                              hipStream_t stream) {
    const float*     x     = (const float*)d_in[0];
    const long long* ei    = (const long long*)d_in[1];   // [2, E] int64
    const long long* batch = (const long long*)d_in[2];   // [N]    int64
    const float* Wl[4] = { (const float*)d_in[3], (const float*)d_in[5],
                           (const float*)d_in[7], (const float*)d_in[9] };
    const float* bl[4] = { (const float*)d_in[4], (const float*)d_in[6],
                           (const float*)d_in[8], (const float*)d_in[10] };
    const float* fc1_w = (const float*)d_in[11];
    const float* fc1_b = (const float*)d_in[12];
    const float* fc2_w = (const float*)d_in[13];
    const float* fc2_b = (const float*)d_in[14];
    const float* out_w = (const float*)d_in[15];
    const float* out_b = (const float*)d_in[16];
    float* out = (float*)d_out;

    // workspace layout (floats)
    float* ws   = (float*)d_ws;
    float* dis  = ws;                       // N (padded to 50176)
    float* bufA = dis + 50176;              // N*256  (hlin)
    float* bufB = bufA + (size_t)NN * DD;   // N*256  (agg / layer output f32)
    float* gmax = bufB + (size_t)NN * DD;   // G*256
    float* gsum = gmax + GG * DD;           // G*256
    float* cnt  = gsum + GG * DD;           // G (padded 256)
    float* gcat = cnt + 256;                // G*512
    float* a1   = gcat + GG * 2 * DD;       // G*512
    float* a2   = a1 + GG * 512;            // G*256
    _Float16* bswz = (_Float16*)(a2 + GG * DD);       // K*256 halves (max 64K)
    _Float16* a16  = (_Float16*)(a2 + GG * DD + 32768); // N*256 halves (GEMM input)

    // 1) degrees -> dis = rsqrt(1+deg)
    deg_init_kernel<<<(NN + 255) / 256, 256, 0, stream>>>(dis, NN);
    deg_accum_kernel<<<(EE + 255) / 256, 256, 0, stream>>>(dis, ei, EE);
    deg_finalize_kernel<<<(NN + 255) / 256, 256, 0, stream>>>(dis, NN);

    // layer-0 input x -> f16
    cvt16_kernel<<<(NN * FF / 4 + 255) / 256, 256, 0, stream>>>(x, a16, NN * FF / 4);

    // 2) four GCN layers
    dim3 grid((NN + TM - 1) / TM, DD / TN);
    for (int l = 0; l < 4; ++l) {
        int K = (l == 0) ? FF : DD;
        wswz_kernel<<<(K * DD + 255) / 256, 256, 0, stream>>>(Wl[l], bswz, K);
        if (l == 0)
            gemm_wmma_kernel<FF><<<grid, 256, 0, stream>>>(a16, bswz, bufA, NN);
        else
            gemm_wmma_kernel<DD><<<grid, 256, 0, stream>>>(a16, bswz, bufA, NN);
        agg_init_kernel<<<(NN * DD + 255) / 256, 256, 0, stream>>>(bufA, bufB, dis, NN);
        scatter_kernel<<<(EE + 7) / 8, 256, 0, stream>>>(bufA, bufB, dis, ei, EE);
        bias_tanh_kernel<<<(NN * DD + 255) / 256, 256, 0, stream>>>(bufB, a16, bl[l], NN);
    }

    // 3) pooling
    pool_init_kernel<<<(2 * GG * DD + GG + 255) / 256, 256, 0, stream>>>(gmax, gsum, cnt);
    pool_kernel<<<NN, DD, 0, stream>>>(bufB, batch, gmax, gsum, cnt);
    build_g_kernel<<<(GG * 2 * DD + 255) / 256, 256, 0, stream>>>(gmax, gsum, cnt, gcat);

    // 4) MLP head
    fc_kernel<<<(GG * 512 + 255) / 256, 256, 0, stream>>>(gcat, fc1_w, fc1_b, a1, GG, 512, 512, 1);
    fc_kernel<<<(GG * 256 + 255) / 256, 256, 0, stream>>>(a1, fc2_w, fc2_b, a2, GG, 512, 256, 1);
    fc_kernel<<<(GG * KK + 255) / 256, 256, 0, stream>>>(a2, out_w, out_b, out, GG, 256, KK, 0);
}
